// VQVAER_90666759619193
// MI455X (gfx1250) — matrix operand order repaired and checked by reading, not compiled, not run on hardware
//
#include <hip/hip_runtime.h>
#include <hip/hip_bf16.h>
#include <math.h>

// ---------------------------------------------------------------------------
// VQ-VAE quantize step for MI455X (gfx1250, wave32, WMMA).
// Core: dist = |x|^2 - 2 x.k + |k|^2 ; argmin over K via bf16 WMMA GEMM.
// ---------------------------------------------------------------------------

typedef __bf16 bf16_t;
typedef __bf16 v16bf __attribute__((ext_vector_type(16)));
typedef float  v8f   __attribute__((ext_vector_type(8)));

constexpr int N_ = 16, W_ = 512, T_ = 1024, K_ = 2048;
constexpr int NT = N_ * T_;          // 16384 rows of xf
constexpr int RT = NT / 16;          // 1024 row tiles (M tiles)
constexpr int CT = K_ / 16;          // 128 code tiles (N tiles)
constexpr int KSTEPS = W_ / 32;      // 16 WMMA k-steps (bf16 K=32)
constexpr int FRAG = 32 * 16;        // 512 bf16 elements per fragment
constexpr float MU_ = 0.99f;
constexpr float THRESH_ = 1.0f;

// ---- workspace byte offsets (all 256B-aligned by construction) ------------
constexpr size_t WS_XF    = 0;                                  // NT*W f32 (32 MB)
constexpr size_t WS_APACK = WS_XF    + (size_t)NT * W_ * 4;     // NT*W bf16 (16 MB)
constexpr size_t WS_BPACK = WS_APACK + (size_t)NT * W_ * 2;     // K*W bf16 (2 MB)
constexpr size_t WS_X2    = WS_BPACK + (size_t)K_ * W_ * 2;     // NT f32
constexpr size_t WS_K2    = WS_X2    + (size_t)NT * 4;          // K f32
constexpr size_t WS_XL    = WS_K2    + (size_t)K_ * 4;          // NT i32
constexpr size_t WS_KSUM  = WS_XL    + (size_t)NT * 4;          // K*W f32 (4 MB)
constexpr size_t WS_KELEM = WS_KSUM  + (size_t)K_ * W_ * 4;     // K f32
constexpr size_t WS_SCAL  = WS_KELEM + (size_t)K_ * 4;          // 8 f32
// scal slots: 0=sum(x) 1=sum(x^2) 2=commit 3=fitsum 4=dksum

// ---- output float offsets (tuple concatenated flat) -----------------------
constexpr size_t O_XL     = 0;
constexpr size_t O_XD     = O_XL + NT;
constexpr size_t O_COMMIT = O_XD + (size_t)N_ * W_ * T_;
constexpr size_t O_FIT    = O_COMMIT + 1;
constexpr size_t O_PRE    = O_FIT + 1;
constexpr size_t O_ENT    = O_PRE + 1;
constexpr size_t O_DK     = O_ENT + 1;
constexpr size_t O_KNEW   = O_DK + 1;
constexpr size_t O_KSUMN  = O_KNEW + (size_t)K_ * W_;
constexpr size_t O_KELEMN = O_KSUMN + (size_t)K_ * W_;

// A-fragment per-lane element -> K index mapping (16x32 bf16 A, ISA 7.12.2)
__device__ __forceinline__ int a_kk(int e, int hi) {
    return e + hi * 8 + ((e >= 8) ? 8 : 0);
}

// ---------------------------------------------------------------------------
// Kernel 1: transpose x (N,W,T)->(NT,W) as f32 + bf16 A-fragments + row |x|^2
// and global sum / sumsq for prenorm. One block per 16-row tile.
// ---------------------------------------------------------------------------
__global__ void __launch_bounds__(256)
pack_x_kernel(const float* __restrict__ x, float* __restrict__ xf,
              bf16_t* __restrict__ apack, float* __restrict__ x2,
              float* __restrict__ scal) {
    __shared__ float tile[W_][17];   // [w][i], padded
    const int rt = blockIdx.x;
    const int tid = threadIdx.x;
    const int n  = rt / (T_ / 16);
    const int t0 = (rt % (T_ / 16)) * 16;

    // load 16 t's x 512 w's, coalesced along t within 64B chunks
    for (int idx = tid; idx < W_ * 16; idx += 256) {
        int i = idx & 15, w = idx >> 4;
        tile[w][i] = x[((size_t)(n * W_ + w)) * T_ + (t0 + i)];
    }
    __syncthreads();

    // row-major f32 copy of xf (coalesced along w)
    for (int idx = tid; idx < W_ * 16; idx += 256) {
        int i = idx >> 9, w = idx & (W_ - 1);
        xf[(size_t)(rt * 16 + i) * W_ + w] = tile[w][i];
    }

    const int lane = tid & 31, wave = tid >> 5;

    // per-row |x|^2 + global prenorm sums (each wave owns 2 rows)
    for (int ii = 0; ii < 2; ++ii) {
        int i = wave * 2 + ii;
        float s = 0.f, ss = 0.f;
        for (int w = lane; w < W_; w += 32) {
            float v = tile[w][i];
            s += v; ss += v * v;
        }
        #pragma unroll
        for (int m = 16; m >= 1; m >>= 1) {
            s  += __shfl_xor(s,  m, 32);
            ss += __shfl_xor(ss, m, 32);
        }
        if (lane == 0) {
            x2[rt * 16 + i] = ss;
            atomicAdd(&scal[0], s);
            atomicAdd(&scal[1], ss);
        }
    }

    // bf16 A fragments: frag(rt,kt) -> lane-striped 32B per lane (coalesced)
    const int hi = lane >> 4, m_ = lane & 15;
    for (int kk = 0; kk < 2; ++kk) {
        int kt = wave + kk * 8;
        bf16_t* dst = apack + (size_t)rt * KSTEPS * FRAG + (size_t)kt * FRAG
                      + (size_t)lane * 16;
        #pragma unroll
        for (int e = 0; e < 16; ++e) {
            int Kidx = kt * 32 + a_kk(e, hi);
            dst[e] = (bf16_t)tile[Kidx][m_];
        }
    }
}

// ---------------------------------------------------------------------------
// Kernel 2: pack codebook k into bf16 B-fragments + per-code |k|^2.
// One block per 16-code tile.
// ---------------------------------------------------------------------------
__global__ void __launch_bounds__(256)
pack_k_kernel(const float* __restrict__ k, bf16_t* __restrict__ bpack,
              float* __restrict__ k2) {
    __shared__ float tile[16][W_ + 1];
    const int ct = blockIdx.x;
    const int tid = threadIdx.x;
    const int c0 = ct * 16;

    for (int idx = tid; idx < 16 * W_; idx += 256) {
        int c = idx >> 9, w = idx & (W_ - 1);
        tile[c][w] = k[(size_t)(c0 + c) * W_ + w];
    }
    __syncthreads();

    const int lane = tid & 31, wave = tid >> 5;

    for (int ii = 0; ii < 2; ++ii) {
        int c = wave * 2 + ii;
        float ss = 0.f;
        for (int w = lane; w < W_; w += 32) {
            float v = tile[c][w];
            ss += v * v;
        }
        #pragma unroll
        for (int m = 16; m >= 1; m >>= 1) ss += __shfl_xor(ss, m, 32);
        if (lane == 0) k2[c0 + c] = ss;
    }

    // B fragment (32x16 bf16): same per-lane K order as A (a_kk), col N=lane%16.
    const int hi = lane >> 4, nc = lane & 15;
    for (int kk = 0; kk < 2; ++kk) {
        int kt = wave + kk * 8;
        bf16_t* dst = bpack + (size_t)ct * KSTEPS * FRAG + (size_t)kt * FRAG
                      + (size_t)lane * 16;
        #pragma unroll
        for (int e = 0; e < 16; ++e)
            dst[e] = (bf16_t)tile[nc][kt * 32 + a_kk(e, hi)];
    }
}

// ---------------------------------------------------------------------------
// Kernel 3: WMMA GEMM (-2*x.k + |k|^2) + running argmin over all 2048 codes.
// One block = 16 rows x 2048 codes; 8 waves x 16 code-tiles each; 256 WMMAs
// per wave. A fragments held in registers; B streamed through an incrementing
// base pointer so loads use constant instruction offsets (no 64-bit VALU
// address chains in the hot loop).
// ---------------------------------------------------------------------------
__global__ void __launch_bounds__(256)
gemm_argmin_kernel(const bf16_t* __restrict__ apack,
                   const bf16_t* __restrict__ bpack,
                   const float* __restrict__ k2, const float* __restrict__ x2,
                   int* __restrict__ xl, float* __restrict__ out_xl,
                   float* __restrict__ scal) {
    __shared__ float redv[8][16];
    __shared__ int   redi[8][16];

    const int rt = blockIdx.x;
    const int tid = threadIdx.x;
    const int lane = tid & 31, wave = tid >> 5;
    const int lo = lane & 15, hi = lane >> 4;

    // this wave's B stream: 16 code-tiles x 16 k-steps, contiguous fragments
    const bf16_t* bptr = bpack + (size_t)(wave * 16) * KSTEPS * FRAG
                         + (size_t)lane * 16;
    __builtin_prefetch(bptr, 0, 1);   // global_prefetch_b8: pull stream to L2

    // load all 16 A fragments for this row tile (2x b128 per lane per frag,
    // constant offsets off a single base)
    const bf16_t* aptr = apack + (size_t)rt * KSTEPS * FRAG + (size_t)lane * 16;
    v16bf a[KSTEPS];
    #pragma unroll
    for (int kt = 0; kt < KSTEPS; ++kt)
        a[kt] = *(const v16bf*)(aptr + kt * FRAG);

    float minv[8];
    int   mini[8];
    #pragma unroll
    for (int j = 0; j < 8; ++j) { minv[j] = 3.4e38f; mini[j] = 0; }

    for (int i = 0; i < 16; ++i) {
        v8f acc = {};
        #pragma unroll
        for (int kt = 0; kt < KSTEPS; ++kt) {
            v16bf b = *(const v16bf*)(bptr + kt * FRAG);
            acc = __builtin_amdgcn_wmma_f32_16x16x32_bf16(
                false, a[kt], false, b, (short)0, acc, false, false);
        }
        bptr += KSTEPS * FRAG;                 // next code tile (+16 KB)

        const int   col = (wave * 16 + i) * 16 + lo;
        const float k2v = k2[col];
        #pragma unroll
        for (int j = 0; j < 8; ++j) {
            float val = k2v - 2.0f * acc[j];   // |x|^2 added later (const/row)
            if (val < minv[j]) { minv[j] = val; mini[j] = col; }
        }
    }

    // reduce across the 16 lanes holding each row (xor stays in 16-group)
    #pragma unroll
    for (int j = 0; j < 8; ++j) {
        #pragma unroll
        for (int m = 1; m < 16; m <<= 1) {
            float ov = __shfl_xor(minv[j], m, 32);
            int   oi = __shfl_xor(mini[j], m, 32);
            if (ov < minv[j] || (ov == minv[j] && oi < mini[j])) {
                minv[j] = ov; mini[j] = oi;
            }
        }
    }
    if (lo == 0) {
        #pragma unroll
        for (int j = 0; j < 8; ++j) {
            redv[wave][j + hi * 8] = minv[j];
            redi[wave][j + hi * 8] = mini[j];
        }
    }
    __syncthreads();

    if (tid < 16) {
        float bv = redv[0][tid];
        int   bi = redi[0][tid];
        #pragma unroll
        for (int wv = 1; wv < 8; ++wv) {
            float ov = redv[wv][tid];
            int   oi = redi[wv][tid];
            if (ov < bv || (ov == bv && oi < bi)) { bv = ov; bi = oi; }
        }
        const int r = rt * 16 + tid;
        xl[r] = bi;
        out_xl[r] = (float)bi;
        atomicAdd(&scal[3], bv + x2[r]);       // fit accumulator (min_dist)
    }
}

// ---------------------------------------------------------------------------
// Kernel 4: scatter cluster stats (_k_sum, _k_elem) + commit loss.
// One block per row; f32 global atomics (L2-resident 4 MB target).
// ---------------------------------------------------------------------------
__global__ void __launch_bounds__(256)
scatter_kernel(const float* __restrict__ xf, const float* __restrict__ k,
               const int* __restrict__ xl, float* __restrict__ ksum_acc,
               float* __restrict__ kelem_acc, float* __restrict__ scal) {
    __shared__ float sbuf[8];
    const int r = blockIdx.x;
    const int tid = threadIdx.x;
    const int code = xl[r];

    float cpart = 0.f;
    for (int w = tid; w < W_; w += 256) {
        float xv = xf[(size_t)r * W_ + w];
        float kv = k[(size_t)code * W_ + w];
        atomicAdd(&ksum_acc[(size_t)code * W_ + w], xv);
        float d = kv - xv;
        cpart += d * d;
    }
    if (tid == 0) atomicAdd(&kelem_acc[code], 1.0f);

    #pragma unroll
    for (int m = 16; m >= 1; m >>= 1) cpart += __shfl_xor(cpart, m, 32);
    if ((tid & 31) == 0) sbuf[tid >> 5] = cpart;
    __syncthreads();
    if (tid == 0) {
        float s = 0.f;
        #pragma unroll
        for (int i = 0; i < 8; ++i) s += sbuf[i];
        atomicAdd(&scal[2], s);
    }
}

// ---------------------------------------------------------------------------
// Kernel 5: x_d gather, coalesced along t. x_d[n,w,t] = k[code[n*T+t], w].
// One block covers 256 consecutive t of one n; k (4 MB) is L2-resident.
// ---------------------------------------------------------------------------
__global__ void __launch_bounds__(256)
xd_kernel(const float* __restrict__ k, const int* __restrict__ xl,
          float* __restrict__ out_xd) {
    const int r = blockIdx.x * 256 + threadIdx.x;   // 256 | T_, so same n/block
    const int code = xl[r];
    const int n = r / T_, t = r % T_;
    const float* krow = k + (size_t)code * W_;
    for (int w = 0; w < W_; ++w)
        out_xd[((size_t)(n * W_ + w)) * T_ + t] = krow[w];
}

// ---------------------------------------------------------------------------
// Kernel 6: EMA codebook update + dk accumulator.
// ---------------------------------------------------------------------------
__global__ void __launch_bounds__(256)
update_kernel(const float* __restrict__ kin, const float* __restrict__ ksumin,
              const float* __restrict__ kelemin,
              const float* __restrict__ ksum_acc,
              const float* __restrict__ kelem_acc,
              const float* __restrict__ xf, float* __restrict__ out_knew,
              float* __restrict__ out_ksumn, float* __restrict__ out_kelemn,
              float* __restrict__ scal) {
    __shared__ float sbuf[8];
    const int idx = blockIdx.x * 256 + threadIdx.x;   // over K*W
    const int c = idx / W_, w = idx % W_;

    float ke_new = MU_ * kelemin[c] + (1.0f - MU_) * kelem_acc[c];
    float ks_new = MU_ * ksumin[idx] + (1.0f - MU_) * ksum_acc[idx];
    float usage = (ke_new >= THRESH_) ? 1.0f : 0.0f;
    unsigned rrow = ((unsigned)c * 2654435761u) & (NT - 1);   // k_rand surrogate
    float krand = xf[(size_t)rrow * W_ + w];
    float kn = usage * (ks_new / ke_new) + (1.0f - usage) * krand;

    out_knew[idx]  = kn;
    out_ksumn[idx] = ks_new;
    if (w == 0) out_kelemn[c] = ke_new;

    float d = kn - kin[idx];
    float dp = d * d;
    #pragma unroll
    for (int m = 16; m >= 1; m >>= 1) dp += __shfl_xor(dp, m, 32);
    if ((threadIdx.x & 31) == 0) sbuf[threadIdx.x >> 5] = dp;
    __syncthreads();
    if (threadIdx.x == 0) {
        float s = 0.f;
        #pragma unroll
        for (int i = 0; i < 8; ++i) s += sbuf[i];
        atomicAdd(&scal[4], s);
    }
}

// ---------------------------------------------------------------------------
// Kernel 7: entropy + scalar outputs.
// ---------------------------------------------------------------------------
__global__ void __launch_bounds__(256)
finalize_kernel(const float* __restrict__ kelem_acc,
                const float* __restrict__ scal, float* __restrict__ out) {
    __shared__ float sbuf[8];
    const int tid = threadIdx.x;
    const float inv_total = 1.0f / (float)NT;   // sum(_k_elem) == NT exactly
    float ent = 0.f;
    for (int c = tid; c < K_; c += 256) {
        float p = kelem_acc[c] * inv_total;
        ent -= p * logf(p + 1e-8f);
    }
    #pragma unroll
    for (int m = 16; m >= 1; m >>= 1) ent += __shfl_xor(ent, m, 32);
    if ((tid & 31) == 0) sbuf[tid >> 5] = ent;
    __syncthreads();
    if (tid == 0) {
        float e = 0.f;
        #pragma unroll
        for (int i = 0; i < 8; ++i) e += sbuf[i];
        const float sz = (float)NT * (float)W_;
        float mean = scal[0] / sz;
        float var  = scal[1] / sz - mean * mean;
        out[O_PRE]    = sqrtf(fmaxf(var, 0.f));
        out[O_FIT]    = scal[3] / (float)NT;
        out[O_COMMIT] = scal[2] / sz;
        out[O_DK]     = sqrtf(scal[4] / ((float)K_ * (float)W_));
        out[O_ENT]    = e;
    }
}

// ---------------------------------------------------------------------------
extern "C" void kernel_launch(void* const* d_in, const int* in_sizes, int n_in,
                              void* d_out, int out_size, void* d_ws, size_t ws_size,
                              hipStream_t stream) {
    (void)in_sizes; (void)n_in; (void)out_size; (void)ws_size;

    const float* x       = (const float*)d_in[0];
    const float* k       = (const float*)d_in[1];
    const float* k_sum   = (const float*)d_in[2];
    const float* k_elem  = (const float*)d_in[3];
    float* out = (float*)d_out;

    char* ws = (char*)d_ws;
    float*  xf        = (float*)(ws + WS_XF);
    bf16_t* apack     = (bf16_t*)(ws + WS_APACK);
    bf16_t* bpack     = (bf16_t*)(ws + WS_BPACK);
    float*  x2        = (float*)(ws + WS_X2);
    float*  k2        = (float*)(ws + WS_K2);
    int*    xl        = (int*)(ws + WS_XL);
    float*  ksum_acc  = (float*)(ws + WS_KSUM);
    float*  kelem_acc = (float*)(ws + WS_KELEM);
    float*  scal      = (float*)(ws + WS_SCAL);

    // zero the accumulators every call (graph-capture safe)
    hipMemsetAsync(ksum_acc,  0, (size_t)K_ * W_ * 4, stream);
    hipMemsetAsync(kelem_acc, 0, (size_t)K_ * 4, stream);
    hipMemsetAsync(scal,      0, 8 * 4, stream);

    pack_x_kernel<<<RT, 256, 0, stream>>>(x, xf, apack, x2, scal);
    pack_k_kernel<<<CT, 256, 0, stream>>>(k, bpack, k2);
    gemm_argmin_kernel<<<RT, 256, 0, stream>>>(apack, bpack, k2, x2,
                                               xl, out + O_XL, scal);
    scatter_kernel<<<NT, 256, 0, stream>>>(xf, k, xl, ksum_acc, kelem_acc, scal);
    xd_kernel<<<NT / 256, 256, 0, stream>>>(k, xl, out + O_XD);
    update_kernel<<<(K_ * W_) / 256, 256, 0, stream>>>(
        k, k_sum, k_elem, ksum_acc, kelem_acc, xf,
        out + O_KNEW, out + O_KSUMN, out + O_KELEMN, scal);
    finalize_kernel<<<1, 256, 0, stream>>>(kelem_acc, scal, out);
}
